// PerturbModel_25091198943810
// MI455X (gfx1250) — matrix operand order repaired and checked by reading, not compiled, not run on hardware
//
#include <hip/hip_runtime.h>
#include <hip/hip_bf16.h>
#include <math.h>

typedef __bf16 bf16;
typedef __attribute__((ext_vector_type(16))) __bf16 v16bf;
typedef __attribute__((ext_vector_type(8)))  __bf16 v8bf;
typedef __attribute__((ext_vector_type(8)))  float   v8f;
typedef __attribute__((ext_vector_type(4)))  float   f4;

union BfPack { v16bf v; v8bf h[2]; };

#define Bb   8
#define Ll   2048
#define Dd   256
#define Mm   64
#define Vv   8192
#define ROWS (Bb*Ll)   // 16384

// ---------------------------------------------------------------------------
// Prep: split W_conv into 5 shift-matrices Wck[k][m][d] (bf16), and build
// WvRedT[d][m] = (1/5) * sum_k W_v[m][d*5+k]  (bf16, transposed for B-operand)
// ---------------------------------------------------------------------------
__global__ void prep_weights(const float* __restrict__ Wconv,
                             const float* __restrict__ Wv,
                             bf16* __restrict__ Wck,
                             bf16* __restrict__ WvRedT) {
  int idx = blockIdx.x * blockDim.x + threadIdx.x;   // 0 .. 64*256-1
  if (idx >= Mm * Dd) return;
  int m = idx / Dd, d = idx % Dd;
  float s = 0.f;
#pragma unroll
  for (int k = 0; k < 5; k++) {
    float w = Wconv[(size_t)m * (Dd * 5) + d * 5 + k];
    Wck[((size_t)(k * Mm + m)) * Dd + d] = (bf16)w;
    s += Wv[(size_t)m * (Dd * 5) + d * 5 + k];
  }
  WvRedT[(size_t)d * Mm + m] = (bf16)(0.2f * s);
}

// f32 -> (optional f32 copy) + bf16
__global__ void prep_cast(const float* __restrict__ src,
                          float* __restrict__ dstF,
                          bf16* __restrict__ dstB, int n) {
  int i = blockIdx.x * blockDim.x + threadIdx.x;
  if (i >= n) return;
  float v = src[i];
  if (dstF) dstF[i] = v;
  dstB[i] = (bf16)v;
}

// ---------------------------------------------------------------------------
// One descent step.  256 threads = 8 waves = 2 row-tiles x 4 cooperating waves.
// Per tile: wave `sub` owns m-tile `sub` in phase 1 (5*8 = 40 WMMAs) and
// d-tiles [4*sub, 4*sub+4) in phase 2 (4*2 = 8 WMMAs).  Softmax and row-std
// reductions go lane-shfl -> LDS -> cross-wave.
// ---------------------------------------------------------------------------
__global__ __launch_bounds__(256)
void step_kernel(const float* __restrict__ Xin, const bf16* __restrict__ XbfIn,
                 float* __restrict__ Xout, bf16* __restrict__ XbfOut,
                 const bf16* __restrict__ Wck, const float* __restrict__ bconv,
                 const bf16* __restrict__ WvRedT) {
  __shared__ float red[2][4][16][2];      // per-tile / per-sub / per-row / {a,b}
  __shared__ bf16  xp_lds[2][16][Mm];     // 4 KB
  const int wave = threadIdx.x >> 5;
  const int lane = threadIdx.x & 31;
  const int half = lane >> 4;
  const int l16  = lane & 15;
  const int t    = wave >> 2;             // tile slot within block (0/1)
  const int sub  = wave & 3;              // cooperating-wave id
  const int tile = blockIdx.x * 2 + t;
  const int row0 = tile * 16;
  const int b    = row0 >> 11;            // /2048
  const int l0   = row0 & 2047;
  const int myl  = l0 + l16;

  // ---- phase 1: xs 16x16 tile for m in [sub*16, sub*16+16) ----
  v8f acc = {};
  const int offs[5] = {2, 1, 0, -1, -2};  // k=0..4 -> roll(-2..2)
#pragma unroll
  for (int k = 0; k < 5; k++) {
    const int nrow = (b << 11) + ((myl + offs[k] + 2048) & 2047);
    const bf16* arow = XbfIn + (size_t)nrow * Dd;
#pragma unroll
    for (int kb = 0; kb < Dd; kb += 32) {
      BfPack a;                            // A: 16x32, row = l16
      a.h[0] = *(const v8bf*)(arow + kb + half * 16);
      a.h[1] = *(const v8bf*)(arow + kb + half * 16 + 8);
      const bf16* brow = Wck + ((size_t)(k * Mm + sub * 16 + l16)) * Dd + kb + half * 16;
      BfPack bm;                           // B: 32x16, col = m
      bm.h[0] = *(const v8bf*)(brow);
      bm.h[1] = *(const v8bf*)(brow + 8);
      acc = __builtin_amdgcn_wmma_f32_16x16x32_bf16(
          false, a.v, false, bm.v, (short)0, acc, false, false);
    }
  }
  {
    float bc = bconv[sub * 16 + l16];
#pragma unroll
    for (int r = 0; r < 8; r++) acc[r] += bc;
  }
  // wave-local per-row max -> LDS
#pragma unroll
  for (int r = 0; r < 8; r++) {
    float m = acc[r];
#pragma unroll
    for (int msk = 1; msk < 16; msk <<= 1) m = fmaxf(m, __shfl_xor(m, msk, 32));
    if (l16 == 0) red[t][sub][r + 8 * half][0] = m;
  }
  __syncthreads();
  // global row max, exp, wave-local sum -> LDS
#pragma unroll
  for (int r = 0; r < 8; r++) {
    const int row = r + 8 * half;
    float m = fmaxf(fmaxf(red[t][0][row][0], red[t][1][row][0]),
                    fmaxf(red[t][2][row][0], red[t][3][row][0]));
    acc[r] = __expf(acc[r] - m);
    float s = acc[r];
#pragma unroll
    for (int msk = 1; msk < 16; msk <<= 1) s += __shfl_xor(s, msk, 32);
    if (l16 == 0) red[t][sub][row][1] = s;
  }
  __syncthreads();
  // normalize, stage xp (bf16) to LDS in [row][m] layout
#pragma unroll
  for (int r = 0; r < 8; r++) {
    const int row = r + 8 * half;
    float s = red[t][0][row][1] + red[t][1][row][1] +
              red[t][2][row][1] + red[t][3][row][1];
    xp_lds[t][row][sub * 16 + l16] = (bf16)(acc[r] / s);
  }
  __syncthreads();

  // ---- phase 2: dx for d-tiles ct = sub*4 .. sub*4+3 ----
  float yv[4][8];
  float sumY[8], sumQ[8];
#pragma unroll
  for (int r = 0; r < 8; r++) { sumY[r] = 0.f; sumQ[r] = 0.f; }
#pragma unroll
  for (int cc = 0; cc < 4; cc++) {
    const int ct = sub * 4 + cc;
    v8f a2 = {};
#pragma unroll
    for (int kb = 0; kb < Mm; kb += 32) {
      BfPack a;
      a.h[0] = *(const v8bf*)(&xp_lds[t][l16][kb + half * 16]);
      a.h[1] = *(const v8bf*)(&xp_lds[t][l16][kb + half * 16 + 8]);
      const bf16* brow = WvRedT + (size_t)(ct * 16 + l16) * Mm + kb + half * 16;
      BfPack bm;
      bm.h[0] = *(const v8bf*)(brow);
      bm.h[1] = *(const v8bf*)(brow + 8);
      a2 = __builtin_amdgcn_wmma_f32_16x16x32_bf16(
          false, a.v, false, bm.v, (short)0, a2, false, false);
    }
    const int d = ct * 16 + l16;
#pragma unroll
    for (int r = 0; r < 8; r++) {
      int grow = row0 + r + 8 * half;
      float y = Xin[(size_t)grow * Dd + d] - 0.1f * a2[r];
      yv[cc][r] = y;
      sumY[r] += y;
      sumQ[r] += y * y;
    }
  }
  // wave-local stat partials -> LDS
#pragma unroll
  for (int r = 0; r < 8; r++) {
    float sy = sumY[r], sq = sumQ[r];
#pragma unroll
    for (int msk = 1; msk < 16; msk <<= 1) {
      sy += __shfl_xor(sy, msk, 32);
      sq += __shfl_xor(sq, msk, 32);
    }
    if (l16 == 0) { red[t][sub][r + 8 * half][0] = sy; red[t][sub][r + 8 * half][1] = sq; }
  }
  __syncthreads();
  float scal[8];
#pragma unroll
  for (int r = 0; r < 8; r++) {
    const int row = r + 8 * half;
    float sy = red[t][0][row][0] + red[t][1][row][0] + red[t][2][row][0] + red[t][3][row][0];
    float sq = red[t][0][row][1] + red[t][1][row][1] + red[t][2][row][1] + red[t][3][row][1];
    float mean = sy * (1.f / 256.f);
    float var  = fmaxf((sq - 256.f * mean * mean) * (1.f / 255.f), 0.f);
    scal[r] = 1.f / (1e-5f + sqrtf(var));
  }
#pragma unroll
  for (int cc = 0; cc < 4; cc++) {
    const int d = (sub * 4 + cc) * 16 + l16;
#pragma unroll
    for (int r = 0; r < 8; r++) {
      int grow = row0 + r + 8 * half;
      float o = yv[cc][r] * scal[r];
      Xout[(size_t)grow * Dd + d]   = o;      // stays hot in L2 (24 MB set)
      XbfOut[(size_t)grow * Dd + d] = (bf16)o;
    }
  }
}

// ---------------------------------------------------------------------------
// logits = xr @ embed^T ; software-pipelined K-loop, NT stores for the 512 MB
// stream; per-(row,chunk) max / sumexp partials for the later log-softmax.
// ---------------------------------------------------------------------------
__global__ __launch_bounds__(256)
void logits_kernel(const bf16* __restrict__ Xbf, const bf16* __restrict__ Ebf,
                   float* __restrict__ out, float* __restrict__ pmax,
                   float* __restrict__ psum) {
  __shared__ float smax[8][16];
  __shared__ float ssum[8][16];
  const int wave = threadIdx.x >> 5, lane = threadIdx.x & 31;
  const int half = lane >> 4, l16 = lane & 15;
  const int rowTile = blockIdx.x >> 3;
  const int chunk   = blockIdx.x & 7;
  const int row0    = rowTile * 16;
  const int c0      = chunk * 1024 + wave * 128;

  const bf16* arow = Xbf + (size_t)(row0 + l16) * Dd;
  const bf16* bbase = Ebf + (size_t)(c0 + l16) * Dd + half * 16;

  v8f acc[8] = {{}, {}, {}, {}, {}, {}, {}, {}};
  BfPack a_cur, b_cur;
  a_cur.h[0] = *(const v8bf*)(arow + half * 16);
  a_cur.h[1] = *(const v8bf*)(arow + half * 16 + 8);
  b_cur.h[0] = *(const v8bf*)(bbase);
  b_cur.h[1] = *(const v8bf*)(bbase + 8);
#pragma unroll
  for (int kb = 0; kb < 8; kb++) {              // K chunk = kb*32
    BfPack a_nxt = a_cur;
    if (kb < 7) {
      a_nxt.h[0] = *(const v8bf*)(arow + (kb + 1) * 32 + half * 16);
      a_nxt.h[1] = *(const v8bf*)(arow + (kb + 1) * 32 + half * 16 + 8);
    }
#pragma unroll
    for (int ct = 0; ct < 8; ct++) {
      BfPack b_nxt = b_cur;
      const int nkb = (ct < 7) ? kb : kb + 1;
      const int nct = (ct < 7) ? ct + 1 : 0;
      if (nkb < 8) {                             // prefetch next B tile
        const bf16* bp = bbase + (size_t)nct * 16 * Dd + nkb * 32;
        b_nxt.h[0] = *(const v8bf*)(bp);
        b_nxt.h[1] = *(const v8bf*)(bp + 8);
      }
      acc[ct] = __builtin_amdgcn_wmma_f32_16x16x32_bf16(
          false, a_cur.v, false, b_cur.v, (short)0, acc[ct], false, false);
      b_cur = b_nxt;
    }
    a_cur = a_nxt;
  }
  // wave-local row max -> LDS
  float mx[8];
#pragma unroll
  for (int r = 0; r < 8; r++) {
    float m = acc[0][r];
#pragma unroll
    for (int ct = 1; ct < 8; ct++) m = fmaxf(m, acc[ct][r]);
#pragma unroll
    for (int msk = 1; msk < 16; msk <<= 1) m = fmaxf(m, __shfl_xor(m, msk, 32));
    mx[r] = m;
  }
  if (l16 == 0) {
#pragma unroll
    for (int r = 0; r < 8; r++) smax[wave][r + 8 * half] = mx[r];
  }
  __syncthreads();
  // chunk max; wave-local sumexp vs chunk max
  float sm[8];
#pragma unroll
  for (int r = 0; r < 8; r++) {
    float cm = smax[0][r + 8 * half];
#pragma unroll
    for (int w = 1; w < 8; w++) cm = fmaxf(cm, smax[w][r + 8 * half]);
    float s = 0.f;
#pragma unroll
    for (int ct = 0; ct < 8; ct++) s += __expf(acc[ct][r] - cm);
#pragma unroll
    for (int msk = 1; msk < 16; msk <<= 1) s += __shfl_xor(s, msk, 32);
    sm[r] = s;
  }
  if (l16 == 0) {
#pragma unroll
    for (int r = 0; r < 8; r++) ssum[wave][r + 8 * half] = sm[r];
  }
  // raw logits out (non-temporal: 512 MB stream, don't thrash L2)
#pragma unroll
  for (int ct = 0; ct < 8; ct++)
#pragma unroll
    for (int r = 0; r < 8; r++)
      __builtin_nontemporal_store(
          acc[ct][r],
          &out[(size_t)(row0 + r + 8 * half) * Vv + c0 + ct * 16 + l16]);
  __syncthreads();
  if (wave == 0 && lane < 16) {
    int row = row0 + lane;
    float m = smax[0][lane], s = ssum[0][lane];
#pragma unroll
    for (int w = 1; w < 8; w++) { m = fmaxf(m, smax[w][lane]); s += ssum[w][lane]; }
    pmax[(size_t)row * 8 + chunk] = m;
    psum[(size_t)row * 8 + chunk] = s;
  }
}

__global__ void lse_kernel(const float* __restrict__ pmax,
                           const float* __restrict__ psum,
                           float* __restrict__ lse) {
  int row = blockIdx.x * blockDim.x + threadIdx.x;
  if (row >= ROWS) return;
  float m = pmax[(size_t)row * 8];
#pragma unroll
  for (int c = 1; c < 8; c++) m = fmaxf(m, pmax[(size_t)row * 8 + c]);
  float s = 0.f;
#pragma unroll
  for (int c = 0; c < 8; c++) s += psum[(size_t)row * 8 + c] * __expf(pmax[(size_t)row * 8 + c] - m);
  lse[row] = m + logf(s);
}

__global__ void sub_kernel(float* __restrict__ out, const float* __restrict__ lse) {
  size_t i = (size_t)blockIdx.x * blockDim.x + threadIdx.x;   // float4 index
  f4* o4 = (f4*)out;
  int row = (int)(i >> 11);                                   // (i*4)/8192
  f4 v = __builtin_nontemporal_load(o4 + i);
  float l = lse[row];
  v[0] -= l; v[1] -= l; v[2] -= l; v[3] -= l;
  __builtin_nontemporal_store(v, o4 + i);
}

// ---------------------------------------------------------------------------
extern "C" void kernel_launch(void* const* d_in, const int* in_sizes, int n_in,
                              void* d_out, int out_size, void* d_ws, size_t ws_size,
                              hipStream_t stream) {
  (void)in_sizes; (void)n_in; (void)ws_size;
  const float* xc    = (const float*)d_in[0];
  const float* Wconv = (const float*)d_in[1];
  const float* bconv = (const float*)d_in[2];
  const float* Wv    = (const float*)d_in[3];
  const float* embed = (const float*)d_in[4];

  char* ws = (char*)d_ws;
  size_t off = 0;
  auto carve = [&](size_t bytes) -> char* {
    char* p = ws + off;
    off += (bytes + 255) & ~(size_t)255;
    return p;
  };
  float* X0     = (float*)carve((size_t)ROWS * Dd * 4);
  float* X1     = (float*)carve((size_t)ROWS * Dd * 4);
  bf16*  Xb0    = (bf16*)carve((size_t)ROWS * Dd * 2);
  bf16*  Xb1    = (bf16*)carve((size_t)ROWS * Dd * 2);
  bf16*  Ebf    = (bf16*)carve((size_t)Vv * Dd * 2);
  bf16*  Wck    = (bf16*)carve((size_t)5 * Mm * Dd * 2);
  bf16*  WvRedT = (bf16*)carve((size_t)Dd * Mm * 2);
  float* pmax   = (float*)carve((size_t)ROWS * 8 * 4);
  float* psum   = (float*)carve((size_t)ROWS * 8 * 4);
  float* lse    = (float*)carve((size_t)ROWS * 4);

  prep_weights<<<(Mm * Dd + 255) / 256, 256, 0, stream>>>(Wconv, Wv, Wck, WvRedT);
  prep_cast<<<(ROWS * Dd + 255) / 256, 256, 0, stream>>>(xc, X0, Xb0, ROWS * Dd);
  prep_cast<<<(Vv * Dd + 255) / 256, 256, 0, stream>>>(embed, nullptr, Ebf, Vv * Dd);

  float* Xi = X0; float* Xo = X1;
  bf16*  Bi = Xb0; bf16*  Bo = Xb1;
  for (int s = 0; s < 20; s++) {
    step_kernel<<<ROWS / 16 / 2, 256, 0, stream>>>(Xi, Bi, Xo, Bo, Wck, bconv, WvRedT);
    float* tf = Xi; Xi = Xo; Xo = tf;
    bf16*  tb = Bi; Bi = Bo; Bo = tb;
  }

  logits_kernel<<<(ROWS / 16) * 8, 256, 0, stream>>>(Bi, Ebf, (float*)d_out, pmax, psum);
  lse_kernel<<<(ROWS + 255) / 256, 256, 0, stream>>>(pmax, psum, lse);
  sub_kernel<<<(size_t)out_size / 4 / 256, 256, 0, stream>>>((float*)d_out, lse);
}
